// MultiheadAttention_16793322127856
// MI455X (gfx1250) — matrix operand order repaired and checked by reading, not compiled, not run on hardware
//
#include <hip/hip_runtime.h>
#include <hip/hip_bf16.h>

// ---------- types ----------
typedef __attribute__((ext_vector_type(16))) __bf16        v16bf;
typedef __attribute__((ext_vector_type(8)))  float         v8f;
typedef __attribute__((ext_vector_type(4)))  float         v4f;
typedef __attribute__((ext_vector_type(4)))  unsigned int  v4u;
typedef __attribute__((ext_vector_type(2)))  unsigned int  v2u;
typedef __attribute__((ext_vector_type(8)))  int           v8i;
typedef __attribute__((ext_vector_type(4)))  int           v4i;
typedef unsigned short     u16;
typedef unsigned int       u32;
typedef unsigned long long u64;

#define BDIM 4
#define SDIM 2048
#define DDIM 1024
#define HDIM 16
#define EDIM 64
#define LDSP 40   // LDS row pitch (bf16 elems): 80B rows keep 16B alignment

union Frag {
  v16bf bf;
  v4u   q[2];
  u16   h[16];
};

__device__ __forceinline__ u16 f2bf(float f) {
  u32 u = __builtin_bit_cast(u32, f);
  u32 r = u + 0x7FFFu + ((u >> 16) & 1u);   // round-to-nearest-even
  return (u16)(r >> 16);
}

__device__ __forceinline__ v8f wmma_bf16(v16bf a, v16bf b, v8f c) {
  return __builtin_amdgcn_wmma_f32_16x16x32_bf16(false, a, false, b, (short)0, c,
                                                 false, false);
}

__device__ __forceinline__ float redmax16(float v) {
  v = fmaxf(v, __shfl_xor(v, 1, 32));
  v = fmaxf(v, __shfl_xor(v, 2, 32));
  v = fmaxf(v, __shfl_xor(v, 4, 32));
  v = fmaxf(v, __shfl_xor(v, 8, 32));
  return v;
}
__device__ __forceinline__ float redsum16(float v) {
  v += __shfl_xor(v, 1, 32);
  v += __shfl_xor(v, 2, 32);
  v += __shfl_xor(v, 4, 32);
  v += __shfl_xor(v, 8, 32);
  return v;
}

// Low 32 bits of a flat shared-aperture address = LDS byte offset (ISA 10.2).
__device__ __forceinline__ u32 lds_off(const void* p) {
  return (u32)(uintptr_t)p;
}

// ---------- TDM: 2-D bf16 tile load, D# per CDNA5 ISA ----
// clang-23 builtin form (6 args):
//   (uint32x4 g0, int32x8 g1, int32x4 g2, int32x4 g3, int32x8 g4, i32 cpol)
__device__ __forceinline__ void tdm_load_2d_bf16(u32 lds_addr, const void* gtile,
                                                 u32 tile_d0, u32 tile_d1,
                                                 u32 tens_d0, u32 tens_d1,
                                                 u32 stride0, u32 g1w0_extra) {
  u64 ga = (u64)(uintptr_t)gtile;
  v4u g0;
  g0[0] = 1u;                                            // count=1, user D#
  g0[1] = lds_addr;                                      // lds_addr (bytes)
  g0[2] = (u32)ga;                                       // global_addr[31:0]
  g0[3] = (u32)((ga >> 32) & 0x1FFFFFFu) | (2u << 30);   // ga[56:32] | type=2
  v8i g1;
  g1[0] = (int)(0x00010000u | g1w0_extra);               // data_size=1 (2B)
  g1[1] = (int)((tens_d0 & 0xFFFFu) << 16);              // tensor_dim0 lo16
  g1[2] = (int)(((tens_d0 >> 16) & 0xFFFFu) | ((tens_d1 & 0xFFFFu) << 16));
  g1[3] = (int)(((tens_d1 >> 16) & 0xFFFFu) | ((tile_d0 & 0xFFFFu) << 16));
  g1[4] = (int)(tile_d1 & 0xFFFFu);                      // tile_dim1, tile_dim2=0
  g1[5] = (int)stride0;                                  // tensor_dim0_stride lo32
  g1[6] = 0;
  g1[7] = 0;
  v4i g2 = {0, 0, 0, 0};                                 // 2-D: groups 2/3 unused
  v4i g3 = {0, 0, 0, 0};
  v8i g4 = {0, 0, 0, 0, 0, 0, 0, 0};                     // extension group: zeros
  __builtin_amdgcn_tensor_load_to_lds(g0, g1, g2, g3, g4, 0);
}
// pad_enable | pad_interval=16 DWORD (code 3) | pad_amount=4 DWORD (code 3):
// 64B data rows + 16B pad -> LDS pitch 40 bf16 == LDSP.
#define TDM_PAD40 ((1u << 20) | (3u << 22) | (3u << 25))

// ---------- kernel 1: fp32 -> bf16 (4 elems/thread) ----------
__global__ void cvt_bf16_kernel(const float* __restrict__ in,
                                u16* __restrict__ out, int n4) {
  int i = blockIdx.x * blockDim.x + threadIdx.x;
  if (i >= n4) return;
  v4f x = ((const v4f*)in)[i];
  v2u p;
  p.x = (u32)f2bf(x.x) | ((u32)f2bf(x.y) << 16);
  p.y = (u32)f2bf(x.z) | ((u32)f2bf(x.w) << 16);
  ((v2u*)out)[i] = p;
}

// ---------- kernel 2: QKV projection GEMM ----------
// xbf: [8192][1024] bf16 ; wbf: [3][H][D][E] bf16 ; qkv: [3][B][H][S][E] bf16
// grid (32, 16, 3), block 256 (8 waves). Wave owns 32x64 tile (2 M-halves
// sharing B fragments). x operand streamed per-wave via double-buffered TDM.
__global__ void __launch_bounds__(256)
qkv_gemm_kernel(const u16* __restrict__ xbf, const u16* __restrict__ wbf,
                u16* __restrict__ qkv) {
  __shared__ __align__(16) u16 Wt[EDIM * LDSP];          // W chunk ^T [e][k]
  __shared__ __align__(16) u16 Xt[8][2][32 * LDSP];      // per-wave x dbl-buf

  const int mat    = blockIdx.z;
  const int h      = blockIdx.y;
  const int rowblk = blockIdx.x * 256;
  const int tid    = threadIdx.x;
  const int wv     = __builtin_amdgcn_readfirstlane(tid >> 5);  // uniform wave id
  const int lane   = tid & 31;
  const int ln     = lane & 15;
  const int hl     = lane >> 4;

  const u16* wsrc = wbf + ((size_t)(mat * HDIM + h)) * DDIM * EDIM;
  const u16* gx   = xbf + (size_t)(rowblk + wv * 32) * DDIM;  // wave's 32 rows

  v8f zero = {0, 0, 0, 0, 0, 0, 0, 0};
  v8f acc[2][4];
#pragma unroll
  for (int mt = 0; mt < 2; ++mt)
#pragma unroll
    for (int nt = 0; nt < 4; ++nt) acc[mt][nt] = zero;

  const int sd = tid & 31;        // staging: k index within chunk
  const int se = (tid >> 5) * 8;  // staging: e base

  // prologue: TDM chunk 0 into buffer 0 (one issue per wave; EXEC ignored)
  tdm_load_2d_bf16(lds_off(&Xt[wv][0][0]), gx, 32, 32, DDIM,
                   (u32)(BDIM * SDIM), DDIM, TDM_PAD40);

  for (int c = 0; c < DDIM / 32; ++c) {
    const int kb = c * 32;
    __syncthreads();  // fences all prior LDS traffic incl. reads of both bufs

    // issue next x chunk into the other buffer (last iter: harmless dup)
    const int kn = (c + 1 < DDIM / 32) ? kb + 32 : kb;
    tdm_load_2d_bf16(lds_off(&Xt[wv][(c + 1) & 1][0]), gx + kn, 32, 32, DDIM,
                     (u32)(BDIM * SDIM), DDIM, TDM_PAD40);

    {  // stage W[kb..kb+31][0..63] transposed into LDS (all 256 threads)
      v4u w = *(const v4u*)&wsrc[(size_t)(kb + sd) * EDIM + se];
      u16 tmp[8];
      *(v4u*)tmp = w;
#pragma unroll
      for (int i = 0; i < 8; ++i) Wt[(se + i) * LDSP + sd] = tmp[i];
    }
    __syncthreads();
    __builtin_amdgcn_s_wait_tensorcnt(1);  // chunk c resident; chunk c+1 in flight

    const u16* xt = &Xt[wv][c & 1][0];
    Frag a[2];
#pragma unroll
    for (int mt = 0; mt < 2; ++mt) {  // A: lane row = mt*16+ln; k = 8*hl+{0..7,16..23}
      a[mt].q[0] = *(const v4u*)&xt[(mt * 16 + ln) * LDSP + 8 * hl];
      a[mt].q[1] = *(const v4u*)&xt[(mt * 16 + ln) * LDSP + 8 * hl + 16];
    }
#pragma unroll
    for (int nt = 0; nt < 4; ++nt) {  // B: lane col = 16*nt+ln; k = 8*hl+{0..7,16..23}
      Frag b;
      b.q[0] = *(const v4u*)&Wt[(nt * 16 + ln) * LDSP + 8 * hl];
      b.q[1] = *(const v4u*)&Wt[(nt * 16 + ln) * LDSP + 8 * hl + 16];
      acc[0][nt] = wmma_bf16(a[0].bf, b.bf, acc[0][nt]);
      acc[1][nt] = wmma_bf16(a[1].bf, b.bf, acc[1][nt]);
    }
  }

  const size_t matstride = (size_t)BDIM * HDIM * SDIM * EDIM;
#pragma unroll
  for (int mt = 0; mt < 2; ++mt)
#pragma unroll
    for (int nt = 0; nt < 4; ++nt)
#pragma unroll
      for (int r = 0; r < 8; ++r) {
        int grow = rowblk + wv * 32 + mt * 16 + r + 8 * hl;
        int bb   = grow >> 11;
        int ss   = grow & 2047;
        size_t dst = (size_t)mat * matstride +
                     (((size_t)(bb * HDIM + h)) * SDIM + ss) * EDIM + nt * 16 + ln;
        qkv[dst] = f2bf(acc[mt][nt][r]);
      }
}

// ---------- kernel 3: causal flash attention ----------
// grid (S/128=16, H, B), block 128 (4 waves); wave owns 32 query rows
// (2 M-halves), K/V fragments reused across both halves.
__global__ void __launch_bounds__(128)
attn_kernel(const u16* __restrict__ qkv, float* __restrict__ out) {
  __shared__ __align__(16) u16 Vt[EDIM * LDSP];    // V tile ^T [e][key]
  __shared__ __align__(16) u16 Pt[4][32 * LDSP];   // per-wave P tile [row][key]

  const int b     = blockIdx.z;
  const int h     = blockIdx.y;
  const int qbase = blockIdx.x * 128;
  const int tid   = threadIdx.x;
  const int wave  = tid >> 5;
  const int lane  = tid & 31;
  const int ln    = lane & 15;
  const int hl    = lane >> 4;

  const size_t matstride = (size_t)BDIM * HDIM * SDIM * EDIM;
  const size_t bh        = (((size_t)b * HDIM) + h) * SDIM * EDIM;
  const u16* qbf = qkv + bh;
  const u16* kbf = qkv + matstride + bh;
  const u16* vbf = qkv + 2 * matstride + bh;

  // Q fragments: 2 M-halves x 2 E-chunks
  Frag aq[2][2];
#pragma unroll
  for (int mt = 0; mt < 2; ++mt) {
    const u16* qrow = qbf + (size_t)(qbase + wave * 32 + mt * 16 + ln) * EDIM;
#pragma unroll
    for (int c = 0; c < 2; ++c) {
      aq[mt][c].q[0] = *(const v4u*)&qrow[c * 32 + 8 * hl];
      aq[mt][c].q[1] = *(const v4u*)&qrow[c * 32 + 8 * hl + 16];
    }
  }

  v8f zero = {0, 0, 0, 0, 0, 0, 0, 0};
  v8f o[2][4];
  float m_i[2][8], l_i[2][8];
#pragma unroll
  for (int mt = 0; mt < 2; ++mt) {
#pragma unroll
    for (int nt = 0; nt < 4; ++nt) o[mt][nt] = zero;
#pragma unroll
    for (int r = 0; r < 8; ++r) { m_i[mt][r] = -1.0e30f; l_i[mt][r] = 0.0f; }
  }

  const int nkt  = qbase / 32 + 4;  // key tiles of 32, through the diagonal
  const int skey = tid >> 2;        // staging: key row 0..31
  const int sev  = (tid & 3) * 16;  // staging: e base 0/16/32/48

  for (int kt = 0; kt < nkt; ++kt) {
    const int kbase = kt * 32;

    __syncthreads();
    {  // stage V[kbase..+31][0..63] transposed into LDS
      const u16* vrow = vbf + (size_t)(kbase + skey) * EDIM + sev;
      v4u w0 = *(const v4u*)&vrow[0];
      v4u w1 = *(const v4u*)&vrow[8];
      u16 tmp[16];
      *(v4u*)&tmp[0] = w0;
      *(v4u*)&tmp[8] = w1;
#pragma unroll
      for (int i = 0; i < 16; ++i) Vt[(sev + i) * LDSP + skey] = tmp[i];
    }
    __syncthreads();

    if (kt + 1 < nkt) {  // -> global_prefetch_b8 for next K/V tiles
      __builtin_prefetch(&kbf[(size_t)(kbase + 32 + ln) * EDIM], 0, 3);
      __builtin_prefetch(&vbf[(size_t)(kbase + 32 + ln) * EDIM], 0, 3);
    }

    // scores: K fragments loaded once, reused by both M-halves (8 WMMA)
    v8f s[2][2];
#pragma unroll
    for (int mt = 0; mt < 2; ++mt) { s[mt][0] = zero; s[mt][1] = zero; }
#pragma unroll
    for (int c = 0; c < 2; ++c) {
      Frag bk0, bk1;  // B = K^T: lane col = key, k slots = e
      const u16* kr0 = kbf + (size_t)(kbase + ln) * EDIM + c * 32 + 8 * hl;
      const u16* kr1 = kbf + (size_t)(kbase + 16 + ln) * EDIM + c * 32 + 8 * hl;
      bk0.q[0] = *(const v4u*)&kr0[0];
      bk0.q[1] = *(const v4u*)&kr0[16];
      bk1.q[0] = *(const v4u*)&kr1[0];
      bk1.q[1] = *(const v4u*)&kr1[16];
#pragma unroll
      for (int mt = 0; mt < 2; ++mt) {
        s[mt][0] = wmma_bf16(aq[mt][c].bf, bk0.bf, s[mt][0]);
        s[mt][1] = wmma_bf16(aq[mt][c].bf, bk1.bf, s[mt][1]);
      }
    }

    // online softmax per M-half; lane holds rows mt*16 + r + 8*hl
    float alpha[2][8];
#pragma unroll
    for (int mt = 0; mt < 2; ++mt) {
#pragma unroll
      for (int r = 0; r < 8; ++r) {
        const int  qg  = qbase + wave * 32 + mt * 16 + r + 8 * hl;
        const bool ok0 = (kbase + ln) <= qg;
        const bool ok1 = (kbase + 16 + ln) <= qg;
        const float v0 = s[mt][0][r] * 0.125f;  // 1/sqrt(64)
        const float v1 = s[mt][1][r] * 0.125f;
        float rm = fmaxf(ok0 ? v0 : -3.0e38f, ok1 ? v1 : -3.0e38f);
        rm = redmax16(rm);
        const float mn = fmaxf(m_i[mt][r], rm);
        const float al = __expf(m_i[mt][r] - mn);
        const float p0 = ok0 ? __expf(v0 - mn) : 0.0f;
        const float p1 = ok1 ? __expf(v1 - mn) : 0.0f;
        const float rs = redsum16(p0 + p1);
        l_i[mt][r]   = l_i[mt][r] * al + rs;
        m_i[mt][r]   = mn;
        alpha[mt][r] = al;
        const int prow = mt * 16 + r + 8 * hl;
        Pt[wave][prow * LDSP + ln]      = f2bf(p0);
        Pt[wave][prow * LDSP + ln + 16] = f2bf(p1);
      }
    }

    // P back as A fragments (same-wave LDS RAW; compiler inserts DScnt waits)
    Frag ap[2];
#pragma unroll
    for (int mt = 0; mt < 2; ++mt) {
      ap[mt].q[0] = *(const v4u*)&Pt[wave][(mt * 16 + ln) * LDSP + 8 * hl];
      ap[mt].q[1] = *(const v4u*)&Pt[wave][(mt * 16 + ln) * LDSP + 8 * hl + 16];
    }

#pragma unroll
    for (int mt = 0; mt < 2; ++mt)
#pragma unroll
      for (int nt = 0; nt < 4; ++nt)
#pragma unroll
        for (int r = 0; r < 8; ++r) o[mt][nt][r] *= alpha[mt][r];

#pragma unroll
    for (int nt = 0; nt < 4; ++nt) {  // V fragments shared by both M-halves
      Frag bv;  // B = V: lane col = e (16*nt+ln), k slots = key (8*hl + {0..7,16..23})
      bv.q[0] = *(const v4u*)&Vt[(nt * 16 + ln) * LDSP + 8 * hl];
      bv.q[1] = *(const v4u*)&Vt[(nt * 16 + ln) * LDSP + 8 * hl + 16];
      o[0][nt] = wmma_bf16(ap[0].bf, bv.bf, o[0][nt]);
      o[1][nt] = wmma_bf16(ap[1].bf, bv.bf, o[1][nt]);
    }
  }

  // epilogue: out[b][s][h*64 + e] = O / l
#pragma unroll
  for (int mt = 0; mt < 2; ++mt)
#pragma unroll
    for (int nt = 0; nt < 4; ++nt)
#pragma unroll
      for (int r = 0; r < 8; ++r) {
        const int srow = qbase + wave * 32 + mt * 16 + r + 8 * hl;
        out[(((size_t)b * SDIM) + srow) * (HDIM * EDIM) + h * EDIM + nt * 16 + ln] =
            o[mt][nt][r] / l_i[mt][r];
      }
}

// ---------- host ----------
extern "C" void kernel_launch(void* const* d_in, const int* in_sizes, int n_in,
                              void* d_out, int out_size, void* d_ws, size_t ws_size,
                              hipStream_t stream) {
  const float* x  = (const float*)d_in[0];
  const float* Wq = (const float*)d_in[1];
  const float* Wk = (const float*)d_in[2];
  const float* Wv = (const float*)d_in[3];
  float* out = (float*)d_out;

  char* ws  = (char*)d_ws;
  u16*  xbf = (u16*)ws;                                   // 16 MiB
  u16*  wbf = (u16*)(ws + (size_t)16 * 1024 * 1024);      //  6 MiB
  u16*  qkv = (u16*)(ws + (size_t)24 * 1024 * 1024);      // 48 MiB (q,k,v)

  const int NX  = BDIM * SDIM * DDIM;        // 8388608
  const int NW  = HDIM * DDIM * EDIM;        // 1048576
  const int NX4 = NX / 4, NW4 = NW / 4;

  cvt_bf16_kernel<<<NX4 / 256, 256, 0, stream>>>(x, xbf, NX4);
  cvt_bf16_kernel<<<NW4 / 256, 256, 0, stream>>>(Wq, wbf, NW4);
  cvt_bf16_kernel<<<NW4 / 256, 256, 0, stream>>>(Wk, wbf + NW, NW4);
  cvt_bf16_kernel<<<NW4 / 256, 256, 0, stream>>>(Wv, wbf + 2 * NW, NW4);

  qkv_gemm_kernel<<<dim3(32, HDIM, 3), 256, 0, stream>>>(xbf, wbf, qkv);
  attn_kernel<<<dim3(SDIM / 128, HDIM, BDIM), 128, 0, stream>>>(qkv, out);
}
